// ProtocolAwareQuantumAttention_57612691309184
// MI455X (gfx1250) — matrix operand order repaired and checked by reading, not compiled, not run on hardware
//
#include <hip/hip_runtime.h>
#include <hip/hip_bf16.h>

typedef __bf16 bf16_t;
typedef __attribute__((ext_vector_type(16))) __bf16 v16bf;
typedef __attribute__((ext_vector_type(8)))  float  v8f;
typedef __attribute__((ext_vector_type(4)))  unsigned int u32x4;
typedef __attribute__((ext_vector_type(8)))  int  i32x8;
typedef __attribute__((ext_vector_type(4)))  int  i32x4;

#define S_LEN 2048
#define DM    1024
#define NH    16
#define DEPTH 64
#define NB    4

static __device__ __forceinline__ v8f wmma_bf16(v16bf a, v16bf b, v8f c) {
  // D = A(16x32 bf16) * B(32x16 bf16) + C(16x16 f32)
  return __builtin_amdgcn_wmma_f32_16x16x32_bf16(false, a, false, b, (short)0, c, false, false);
}

// ---------------------------------------------------------------------------
// TDM: async 2-D bf16 tile load Global -> LDS (Tensor Data Mover, D# per ISA
// CDNA5 ISA).  clang-23 6-arg builtin form.
//   tile_x   : contiguous elements per row  (data_size units, 2B)
//   tile_y   : rows
//   stride_x : row stride in the global tensor (elements)
//   LDS pad  : pad_interval code 3 = every 16 DWORDs (one 64B row),
//              pad_amount  code 3 = 4 DWORDs -> LDS row stride 40 halves.
// ---------------------------------------------------------------------------
static __device__ __forceinline__ void tdm_load_2d_bf16(
    unsigned int lds_off, const bf16_t* gptr,
    unsigned int tile_x, unsigned int tile_y,
    unsigned int tensor_x, unsigned int tensor_y, unsigned int stride_x,
    unsigned int pad_interval_code, unsigned int pad_amount_code, bool pad_en)
{
  const unsigned long long ga = (unsigned long long)(size_t)gptr;
  u32x4 g0;
  g0[0] = 1u;                                    // count=1 (valid, user mode)
  g0[1] = lds_off;                               // lds_addr (bytes)
  g0[2] = (unsigned int)(ga & 0xFFFFFFFFull);    // global_addr lo
  g0[3] = (unsigned int)(ga >> 32) | (2u << 30); // global_addr hi | type=2
  i32x8 g1;
  unsigned int dw0 = (1u << 16);                 // data_size=1 -> 2 bytes
  if (pad_en)
    dw0 |= (1u << 20) | (pad_interval_code << 22) | (pad_amount_code << 25);
  g1[0] = (int)dw0;
  g1[1] = (int)((tensor_x & 0xFFFFu) << 16);                     // tensor_dim0 lo16 @bit48
  g1[2] = (int)((tensor_x >> 16) | ((tensor_y & 0xFFFFu) << 16));// dim0 hi16 | dim1 lo16
  g1[3] = (int)((tensor_y >> 16) | (tile_x << 16));              // dim1 hi16 | tile_dim0
  g1[4] = (int)tile_y;                                           // tile_dim1 (tile_dim2=0)
  g1[5] = (int)stride_x;                                         // tensor_dim0_stride lo32
  g1[6] = 0;                                                     // stride hi16 | dim1_stride lo16
  g1[7] = 0;                                                     // dim1_stride hi32
  i32x4 gz4 = {0, 0, 0, 0};                                      // groups 2/3: 2-D tile
  i32x8 gz8 = {0, 0, 0, 0, 0, 0, 0, 0};                          // extra group (unused)
  __builtin_amdgcn_tensor_load_to_lds(g0, g1, gz4, gz4, gz8, 0);
}

// ---------------------------------------------------------------------------
// Kernel 1: convert + TRANSPOSE the three weight matrices fp32 -> bf16.
//   wT[n][k] = w[k][n]  (so projection B-tiles are row-major for the TDM)
// ---------------------------------------------------------------------------
__global__ __launch_bounds__(256) void cvt3_kernel(
    const float* __restrict__ w0, const float* __restrict__ w1, const float* __restrict__ w2,
    bf16_t* __restrict__ o0, bf16_t* __restrict__ o1, bf16_t* __restrict__ o2)
{
  const float* s;
  bf16_t* d;
  if (blockIdx.z == 0)      { s = w0; d = o0; }
  else if (blockIdx.z == 1) { s = w1; d = o1; }
  else                      { s = w2; d = o2; }
  const int i = (blockIdx.x * 256 + threadIdx.x) * 4;  // 1024 blocks * 256 * 4 = 1M
  const int row = i >> 10;        // k (input dim)
  const int col = i & 1023;       // n (output dim), 4 consecutive
  float4 v = *(const float4*)(s + i);
  d[(size_t)(col + 0) * DM + row] = (bf16_t)v.x;
  d[(size_t)(col + 1) * DM + row] = (bf16_t)v.y;
  d[(size_t)(col + 2) * DM + row] = (bf16_t)v.z;
  d[(size_t)(col + 3) * DM + row] = (bf16_t)v.w;
}

// ---------------------------------------------------------------------------
// Kernel 2: protocol gate  validity[b,h] = sigmoid(pf[b,:] @ wg[:,h] + bg[h])
// ---------------------------------------------------------------------------
__global__ __launch_bounds__(64) void gate_kernel(
    const float* __restrict__ pf, const float* __restrict__ wg,
    const float* __restrict__ bg, float* __restrict__ validity)
{
  const int t = threadIdx.x;            // 0..63
  const int b = t >> 4, h = t & 15;
  float acc = bg[h];
  #pragma unroll 8
  for (int i = 0; i < 64; ++i) acc += pf[b * 64 + i] * wg[i * NH + h];
  validity[t] = 1.0f / (1.0f + __expf(-acc));
}

// ---------------------------------------------------------------------------
// Kernel 3: projection GEMM  X[8192x1024] @ W[1024x1024] + bias
//   z=0: Q -> qh [B,H,S,64]   z=1: K -> kh [B,H,S,64]   z=2: V -> vT [B,H,64,S]
// Block = 256 threads (8 waves). Tile: 128(M) x 64(N), K-step 32.
// W tiles (pre-transposed wT) are DMA'd into LDS by the Tensor Data Mover.
// ---------------------------------------------------------------------------
__global__ __launch_bounds__(256) void proj_kernel(
    const float* __restrict__ q, const float* __restrict__ k, const float* __restrict__ v,
    const bf16_t* __restrict__ wqt, const bf16_t* __restrict__ wkt, const bf16_t* __restrict__ wvt,
    const float* __restrict__ bq, const float* __restrict__ bk, const float* __restrict__ bv,
    bf16_t* __restrict__ qh, bf16_t* __restrict__ kh, bf16_t* __restrict__ vT)
{
  __shared__ bf16_t lA[128][40];   // X tile (bf16), padded rows
  __shared__ bf16_t lB[64][40];    // wT tile [n][k]; TDM padding gives stride 40

  const float* x; const bf16_t* w; const float* bias; bf16_t* out; bool isV = false;
  if (blockIdx.z == 0)      { x = q; w = wqt; bias = bq; out = qh; }
  else if (blockIdx.z == 1) { x = k; w = wkt; bias = bk; out = kh; }
  else                      { x = v; w = wvt; bias = bv; out = vT; isV = true; }

  const int m0 = blockIdx.x * 128;
  const int n0 = blockIdx.y * 64;
  const int tid  = threadIdx.x;
  const int wave = tid >> 5, lane = tid & 31;
  const int c = lane & 15, rh = lane >> 4;

  const int xrow  = tid >> 1;        // 0..127
  const int xcoff = (tid & 1) * 16;  // 0 / 16

  // LDS byte offset of lB (generic shared addr: low 32 bits = LDS offset)
  const unsigned int lb_off = (unsigned int)(size_t)&lB[0][0];

  v8f zero = {};
  v8f acc[4] = {zero, zero, zero, zero};

  for (int kt = 0; kt < 32; ++kt) {
    const int k0 = kt * 32;

    // TDM: async DMA of the 64x32 bf16 W tile (rows n, contiguous k) into lB.
    if (tid < 32) {
      tdm_load_2d_bf16(lb_off, w + (size_t)n0 * DM + k0,
                       /*tile_x=*/32, /*tile_y=*/64,
                       /*tensor_x=*/DM, /*tensor_y=*/DM, /*stride_x=*/DM,
                       /*pad_interval=*/3, /*pad_amount=*/3, /*pad_en=*/true);
    }

    // X tile: 128x32 fp32 -> bf16 LDS (converted in-flight)
    const float* xp = x + (size_t)(m0 + xrow) * DM + k0 + xcoff;
    #pragma unroll
    for (int u = 0; u < 4; ++u) {
      float4 f = *(const float4*)(xp + u * 4);
      lA[xrow][xcoff + u * 4 + 0] = (bf16_t)f.x;
      lA[xrow][xcoff + u * 4 + 1] = (bf16_t)f.y;
      lA[xrow][xcoff + u * 4 + 2] = (bf16_t)f.z;
      lA[xrow][xcoff + u * 4 + 3] = (bf16_t)f.w;
    }
    if (kt < 31) __builtin_prefetch(xp + 32, 0, 1);

    if (tid < 32) __builtin_amdgcn_s_wait_tensorcnt(0);  // TDM done (wave 0)
    __syncthreads();                                     // releases all waves

    // A fragment: row = wave*16 + (lane&15); K chunks (lane>>4)*8 and +16
    v16bf a;
    #pragma unroll
    for (int i = 0; i < 8; ++i) {
      a[i]     = lA[wave * 16 + c][rh * 8 + i];
      a[8 + i] = lA[wave * 16 + c][16 + rh * 8 + i];
    }
    #pragma unroll
    for (int j = 0; j < 4; ++j) {
      // B fragment: col = j*16 + (lane&15); K chunks (lane>>4)*8 and +16 from lB row
      v16bf bm;
      #pragma unroll
      for (int i = 0; i < 16; ++i) bm[i] = lB[j * 16 + c][rh * 8 + i + (i >= 8 ? 8 : 0)];
      acc[j] = wmma_bf16(a, bm, acc[j]);
    }
    __syncthreads();
  }

  // Epilogue: bias add, store bf16 in head-split (or V-transposed) layout
  #pragma unroll
  for (int j = 0; j < 4; ++j) {
    const int n = n0 + j * 16 + c;
    const float bb = bias[n];
    const int h = n >> 6, d = n & 63;
    #pragma unroll
    for (int r = 0; r < 8; ++r) {
      const int m = m0 + wave * 16 + rh * 8 + r;
      const int b = m >> 11, s = m & (S_LEN - 1);
      const float val = acc[j][r] + bb;
      if (!isV) out[((size_t)(b * NH + h) * S_LEN + s) * DEPTH + d] = (bf16_t)val;
      else      out[((size_t)(b * NH + h) * DEPTH + d) * S_LEN + s] = (bf16_t)val;
    }
  }
}

// ---------------------------------------------------------------------------
// Kernel 4: attention. One block per (b, h, 16-row q tile); 8 waves split K.
// Phase 1: logits via WMMA -> attn buffer (raw), online (max,sum) per row.
// Phase 2: cross-wave stat combine in LDS.
// Phase 3: re-read own logits, normalize -> final attn; P.V via WMMA.
// Phase 4: cross-wave O reduction -> concat output.
// ---------------------------------------------------------------------------
__global__ __launch_bounds__(256) void attn_kernel(
    const bf16_t* __restrict__ qh, const bf16_t* __restrict__ kh, const bf16_t* __restrict__ vT,
    const float* __restrict__ validity, const float* __restrict__ mask,
    float* __restrict__ concat, float* __restrict__ attn)
{
  __shared__ bf16_t stage[8][16][32];   // per-wave P tile in bf16 (A-layout source)
  __shared__ float  sM[8][16], sS[8][16];
  __shared__ float  sO[8][16][64];

  const int qt = blockIdx.x & 127;
  const int h  = (blockIdx.x >> 7) & 15;
  const int b  = blockIdx.x >> 11;
  const int bh = b * NH + h;

  const int tid = threadIdx.x, wave = tid >> 5, lane = tid & 31;
  const int c = lane & 15, rh = lane >> 4;

  const float g = validity[bh] * 0.125f;           // fold 1/sqrt(depth)
  const float* maskp = mask + (size_t)b * S_LEN;

  // Q fragments straight from global (native A layout): row = qt*16 + (lane&15)
  const bf16_t* qp = qh + ((size_t)bh * S_LEN + qt * 16 + c) * DEPTH;
  v16bf aq0, aq1;
  #pragma unroll
  for (int i = 0; i < 8; ++i) {
    aq0[i]     = qp[rh * 8 + i];
    aq0[8 + i] = qp[16 + rh * 8 + i];
    aq1[i]     = qp[32 + rh * 8 + i];
    aq1[8 + i] = qp[48 + rh * 8 + i];
  }

  const int kw = wave * 256;          // this wave's K range: [kw, kw+256)
  float runM[8], runS[8];
  #pragma unroll
  for (int r = 0; r < 8; ++r) { runM[r] = -1e30f; runS[r] = 0.0f; }

  // ---- Phase 1 ----
  for (int t = 0; t < 16; ++t) {
    const int p = kw + t * 16 + c;                 // this lane's key column
    const bf16_t* kp = kh + ((size_t)bh * S_LEN + p) * DEPTH;
    if (t < 15) __builtin_prefetch(kp + 16 * DEPTH, 0, 1);
    v16bf b0, b1;                                  // B: col = key pos, depth chunks rh*8 / 16+rh*8
    #pragma unroll
    for (int i = 0; i < 16; ++i) { b0[i] = kp[rh * 8 + i + (i >= 8 ? 8 : 0)]; b1[i] = kp[32 + rh * 8 + i + (i >= 8 ? 8 : 0)]; }
    v8f acc = {};
    acc = wmma_bf16(aq0, b0, acc);
    acc = wmma_bf16(aq1, b1, acc);

    const float madd = maskp[p] * -1.0e9f;
    float l[8];
    float* arow = attn + ((size_t)bh * S_LEN + qt * 16 + rh * 8) * (size_t)S_LEN + p;
    #pragma unroll
    for (int r = 0; r < 8; ++r) {
      l[r] = acc[r] * g + madd;
      arow[(size_t)r * S_LEN] = l[r];              // raw logits into attn output area
    }
    #pragma unroll
    for (int r = 0; r < 8; ++r) {
      float m = l[r];
      m = fmaxf(m, __shfl_xor(m, 1, 32));
      m = fmaxf(m, __shfl_xor(m, 2, 32));
      m = fmaxf(m, __shfl_xor(m, 4, 32));
      m = fmaxf(m, __shfl_xor(m, 8, 32));          // max over the 16 columns (half-wave)
      float e = __expf(l[r] - m);
      e += __shfl_xor(e, 1, 32);
      e += __shfl_xor(e, 2, 32);
      e += __shfl_xor(e, 4, 32);
      e += __shfl_xor(e, 8, 32);                   // sumexp over 16 columns
      const float M2 = fmaxf(runM[r], m);
      runS[r] = runS[r] * __expf(runM[r] - M2) + e * __expf(m - M2);
      runM[r] = M2;
    }
  }

  // ---- Phase 2: combine stats across waves ----
  if (c == 0) {
    #pragma unroll
    for (int r = 0; r < 8; ++r) { sM[wave][rh * 8 + r] = runM[r]; sS[wave][rh * 8 + r] = runS[r]; }
  }
  __syncthreads();

  float gM[8], gInv[8];
  #pragma unroll
  for (int r = 0; r < 8; ++r) {
    const int row = rh * 8 + r;
    float M = -1e30f;
    #pragma unroll
    for (int w = 0; w < 8; ++w) M = fmaxf(M, sM[w][row]);
    float Ssum = 0.0f;
    #pragma unroll
    for (int w = 0; w < 8; ++w) Ssum += sS[w][row] * __expf(sM[w][row] - M);
    gM[r] = M; gInv[r] = 1.0f / Ssum;
  }

  // ---- Phase 3: normalize attn + accumulate O = P @ V ----
  v8f zero = {};
  v8f oacc[4] = {zero, zero, zero, zero};

  for (int step = 0; step < 8; ++step) {
    #pragma unroll
    for (int sub = 0; sub < 2; ++sub) {
      const int t = step * 2 + sub;
      const int p = kw + t * 16 + c;
      float* arow = attn + ((size_t)bh * S_LEN + qt * 16 + rh * 8) * (size_t)S_LEN + p;
      #pragma unroll
      for (int r = 0; r < 8; ++r) {
        const float l  = arow[(size_t)r * S_LEN];  // same-wave readback (HW-ordered)
        const float pr = __expf(l - gM[r]) * gInv[r];
        arow[(size_t)r * S_LEN] = pr;              // final normalized attention
        stage[wave][rh * 8 + r][sub * 16 + c] = (bf16_t)pr;
      }
    }
    // A fragment from staged P: row = lane&15, K chunks rh*8 / 16+rh*8
    v16bf ap;
    #pragma unroll
    for (int i = 0; i < 8; ++i) {
      ap[i]     = stage[wave][c][rh * 8 + i];
      ap[8 + i] = stage[wave][c][16 + rh * 8 + i];
    }
    const int kk0 = kw + step * 32;
    #pragma unroll
    for (int j = 0; j < 4; ++j) {
      // B fragment: col = depth j*16+c ; key chunks rh*8 / 16+rh*8 (vT rows)
      const bf16_t* vp = vT + ((size_t)bh * DEPTH + j * 16 + c) * S_LEN + kk0;
      v16bf bv;
      #pragma unroll
      for (int i = 0; i < 16; ++i) bv[i] = vp[rh * 8 + i + (i >= 8 ? 8 : 0)];
      oacc[j] = wmma_bf16(ap, bv, oacc[j]);
    }
  }

  // ---- Phase 4: reduce partial O across waves, write concat ----
  #pragma unroll
  for (int j = 0; j < 4; ++j)
    #pragma unroll
    for (int r = 0; r < 8; ++r)
      sO[wave][rh * 8 + r][j * 16 + c] = oacc[j][r];
  __syncthreads();

  const int row = tid >> 4;          // 0..15
  const int d0  = (tid & 15) * 4;    // 0..60
  float s0 = 0.f, s1 = 0.f, s2 = 0.f, s3 = 0.f;
  #pragma unroll
  for (int w = 0; w < 8; ++w) {
    s0 += sO[w][row][d0 + 0];
    s1 += sO[w][row][d0 + 1];
    s2 += sO[w][row][d0 + 2];
    s3 += sO[w][row][d0 + 3];
  }
  float* op = concat + ((size_t)b * S_LEN + qt * 16 + row) * DM + h * DEPTH + d0;
  op[0] = s0; op[1] = s1; op[2] = s2; op[3] = s3;
}

// ---------------------------------------------------------------------------
extern "C" void kernel_launch(void* const* d_in, const int* in_sizes, int n_in,
                              void* d_out, int out_size, void* d_ws, size_t ws_size,
                              hipStream_t stream) {
  (void)in_sizes; (void)n_in; (void)out_size; (void)ws_size;
  const float* q    = (const float*)d_in[0];
  const float* k    = (const float*)d_in[1];
  const float* v    = (const float*)d_in[2];
  const float* pf   = (const float*)d_in[3];
  const float* mask = (const float*)d_in[4];
  const float* wq   = (const float*)d_in[5];
  const float* bq   = (const float*)d_in[6];
  const float* wk   = (const float*)d_in[7];
  const float* bk   = (const float*)d_in[8];
  const float* wv   = (const float*)d_in[9];
  const float* bv   = (const float*)d_in[10];
  const float* wg   = (const float*)d_in[11];
  const float* bg   = (const float*)d_in[12];

  const size_t proj_elems = (size_t)NB * NH * S_LEN * DEPTH;  // 8,388,608
  const size_t w_elems    = (size_t)DM * DM;                  // 1,048,576

  bf16_t* qh  = (bf16_t*)d_ws;
  bf16_t* kh  = qh  + proj_elems;
  bf16_t* vT  = kh  + proj_elems;
  bf16_t* wqt = vT  + proj_elems;   // TRANSPOSED bf16 weights [n][k]
  bf16_t* wkt = wqt + w_elems;
  bf16_t* wvt = wkt + w_elems;
  float* validity = (float*)(wvt + w_elems);

  float* concat = (float*)d_out;                              // [B,S,1024]
  float* attnp  = concat + (size_t)NB * S_LEN * DM;           // [B,H,S,S]

  cvt3_kernel<<<dim3(1024, 1, 3), 256, 0, stream>>>(wq, wk, wv, wqt, wkt, wvt);
  gate_kernel<<<1, 64, 0, stream>>>(pf, wg, bg, validity);
  proj_kernel<<<dim3(64, 16, 3), 256, 0, stream>>>(q, k, v, wqt, wkt, wvt,
                                                   bq, bk, bv, qh, kh, vT);
  attn_kernel<<<8192, 256, 0, stream>>>(qh, kh, vT, validity, mask, concat, attnp);
}
